// FermiNetOneElectronLayer_58471684768192
// MI455X (gfx1250) — matrix operand order repaired and checked by reading, not compiled, not run on hardware
//
#include <hip/hip_runtime.h>
#include <math.h>

typedef float v2f __attribute__((ext_vector_type(2)));
typedef float v8f __attribute__((ext_vector_type(8)));
typedef __bf16 v16bf __attribute__((ext_vector_type(16)));
typedef __bf16 v8bf __attribute__((ext_vector_type(8)));
typedef unsigned short v4u16 __attribute__((ext_vector_type(4)));

#define NPART 16
#define D1 256
#define D2 32
#define ND 256
#define KTOT 832
#define BPG 4                 // batches per workgroup
#define MROWS (BPG * NPART)   // 64 rows of the fused activation tile
#define NTHREADS 512          // 16 waves (wave32)

// ---------------- bf16 split helpers (RNE) ----------------
__device__ __forceinline__ float bf2f(unsigned short h) {
  union { unsigned u; float f; } c;
  c.u = ((unsigned)h) << 16;
  return c.f;
}
__device__ __forceinline__ unsigned short f2bf(float x) {
  union { float f; unsigned u; } c;
  c.f = x;
  unsigned r = c.u + 0x7fffu + ((c.u >> 16) & 1u);
  return (unsigned short)(r >> 16);
}
__device__ __forceinline__ void split2(float x, unsigned short& h,
                                       unsigned short& l) {
  h = f2bf(x);
  l = f2bf(x - bf2f(h));
}

// =====================================================================
// bf16x3 path: Xcat (hi/lo bf16) in LDS, weights pre-transposed hi/lo in ws
// =====================================================================
#define SB 840  // bf16 row stride: dword stride 420 -> all-64-banks b128 reads

#define BF_WMMA(ACC, A, Bf)                                                    \
  ACC = __builtin_amdgcn_wmma_f32_16x16x32_bf16(false, (A), false, (Bf),       \
                                                (short)0, ACC, false, false);

__global__ __launch_bounds__(NTHREADS) void ferminet_bf16(
    const float* __restrict__ in_1e,          // (B,16,256)
    const float* __restrict__ in_2e,          // (B,16,16,32)
    const float* __restrict__ bu,             // (256,)
    const unsigned short* __restrict__ wt_hi, // (256,832) K-contiguous bf16
    const unsigned short* __restrict__ wt_lo, // (256,832)
    float* __restrict__ out,                  // (B,16,256)
    int B) {
  extern __shared__ unsigned short sm[];
  unsigned short* Xhi = sm;
  unsigned short* Xlo = sm + MROWS * SB;

  const int tid = threadIdx.x;
  const int lane = tid & 31;
  const int wid = tid >> 5;  // 0..15
  const int b0 = blockIdx.x * BPG;

  // ---- Phase 0: in_1e tile -> Xcat[:,0:256] as hi/lo bf16 ----
  {
#pragma unroll
    for (int it = 0; it < (MROWS * D1) / (NTHREADS * 4); ++it) {
      int idx = (it * NTHREADS + tid) * 4;
      int row = idx >> 8;
      int col = idx & 255;
      int gb = b0 + (row >> 4);
      if (gb > B - 1) gb = B - 1;
      const float4 v = *reinterpret_cast<const float4*>(
          in_1e + ((size_t)gb * NPART + (row & 15)) * D1 + col);
      unsigned short h0, h1, h2, h3, l0, l1, l2, l3;
      split2(v.x, h0, l0); split2(v.y, h1, l1);
      split2(v.z, h2, l2); split2(v.w, h3, l3);
      v4u16 hv = {h0, h1, h2, h3};
      v4u16 lv = {l0, l1, l2, l3};
      *reinterpret_cast<v4u16*>(&Xhi[row * SB + col]) = hv;  // 8B aligned
      *reinterpret_cast<v4u16*>(&Xlo[row * SB + col]) = lv;
    }
  }
  __syncthreads();

  // ---- Phase 1: 1e spin means -> Xcat[:,256:768] (rolled concat) ----
  {
#pragma unroll
    for (int w = 0; w < 4; ++w) {
      int idx = tid + w * NTHREADS;  // (lb, spin t, dim d)
      int lb = idx >> 9;
      int rem = idx & 511;
      int t = rem >> 8;
      int d = rem & 255;
      int base = (lb * NPART + t * 8) * SB + d;
      float s = 0.f;
#pragma unroll
      for (int r = 0; r < 8; ++r)
        s += bf2f(Xhi[base + r * SB]) + bf2f(Xlo[base + r * SB]);
      s *= 0.125f;
      unsigned short hs, ls;
      split2(s, hs, ls);
      int rows_same = (lb * NPART + t * 8) * SB;
      int rows_oth = (lb * NPART + (1 - t) * 8) * SB;
#pragma unroll
      for (int r = 0; r < 8; ++r) {
        Xhi[rows_same + r * SB + 256 + d] = hs;
        Xlo[rows_same + r * SB + 256 + d] = ls;
        Xhi[rows_oth + r * SB + 512 + d] = hs;
        Xlo[rows_oth + r * SB + 512 + d] = ls;
      }
    }
  }

  // ---- Phase 2: 2e column-spin means -> Xcat[:,768:832] (rolled) ----
  {
#pragma unroll
    for (int w = 0; w < 8; ++w) {
      int c = wid + w * 16;  // (lb,m,j); lane = d
      int lb = c >> 5;
      int m = (c >> 1) & 15;
      int j = c & 1;
      int gb = b0 + lb;
      if (gb > B - 1) gb = B - 1;
      const float* p =
          in_2e + (((size_t)gb * NPART + m) * NPART + j * 8) * D2 + lane;
      float s = 0.f;
#pragma unroll
      for (int cc = 0; cc < 8; ++cc) s += p[cc * D2];
      s *= 0.125f;
      unsigned short hs, ls;
      split2(s, hs, ls);
      int srow = lb * NPART + m;
      int sspin = m >> 3;
      int off = srow * SB + 768 + ((j ^ sspin) << 5) + lane;
      Xhi[off] = hs;
      Xlo[off] = ls;
    }
  }
  __syncthreads();

  // ---- Phase 3: bf16x3 GEMM, K = 832 in 26 steps of 32 ----
  const int lh = lane >> 4;
  const int ln = lane & 15;
  const int gcol = (wid << 4) + ln;

  const float bias = bu[gcol];
  v8f acc0, acc1, acc2, acc3;
#pragma unroll
  for (int i = 0; i < 8; ++i) {
    acc0[i] = bias; acc1[i] = bias; acc2[i] = bias; acc3[i] = bias;
  }

  // A per-lane K chunks: {16*lh+0..7} and {16*lh+8..15} (16B each)
  const int ab0 = (0 * 16 + ln) * SB + 16 * lh;
  const int ab1 = ab0 + 16 * SB;
  const int ab2 = ab1 + 16 * SB;
  const int ab3 = ab2 + 16 * SB;
  // B per-lane K chunk: 16*lh + {0..15} (32B contiguous, K-major weights)
  const unsigned short* wh = wt_hi + (size_t)gcol * KTOT + 16 * lh;
  const unsigned short* wl = wt_lo + (size_t)gcol * KTOT + 16 * lh;

#pragma unroll 2
  for (int k = 0; k < KTOT; k += 32) {
    v16bf bhi = *reinterpret_cast<const v16bf*>(wh + k);  // 32B aligned
    v16bf blo = *reinterpret_cast<const v16bf*>(wl + k);

    v8bf h00 = *(const v8bf*)(Xhi + ab0 + k);
    v8bf h01 = *(const v8bf*)(Xhi + ab0 + k + 8);
    v8bf h10 = *(const v8bf*)(Xhi + ab1 + k);
    v8bf h11 = *(const v8bf*)(Xhi + ab1 + k + 8);
    v8bf h20 = *(const v8bf*)(Xhi + ab2 + k);
    v8bf h21 = *(const v8bf*)(Xhi + ab2 + k + 8);
    v8bf h30 = *(const v8bf*)(Xhi + ab3 + k);
    v8bf h31 = *(const v8bf*)(Xhi + ab3 + k + 8);
    v8bf l00 = *(const v8bf*)(Xlo + ab0 + k);
    v8bf l01 = *(const v8bf*)(Xlo + ab0 + k + 8);
    v8bf l10 = *(const v8bf*)(Xlo + ab1 + k);
    v8bf l11 = *(const v8bf*)(Xlo + ab1 + k + 8);
    v8bf l20 = *(const v8bf*)(Xlo + ab2 + k);
    v8bf l21 = *(const v8bf*)(Xlo + ab2 + k + 8);
    v8bf l30 = *(const v8bf*)(Xlo + ab3 + k);
    v8bf l31 = *(const v8bf*)(Xlo + ab3 + k + 8);

#define CAT16(a, b)                                                            \
  __builtin_shufflevector(a, b, 0, 1, 2, 3, 4, 5, 6, 7, 8, 9, 10, 11, 12, 13, \
                          14, 15)
    v16bf ahi0 = CAT16(h00, h01), ahi1 = CAT16(h10, h11);
    v16bf ahi2 = CAT16(h20, h21), ahi3 = CAT16(h30, h31);
    v16bf alo0 = CAT16(l00, l01), alo1 = CAT16(l10, l11);
    v16bf alo2 = CAT16(l20, l21), alo3 = CAT16(l30, l31);
#undef CAT16

    // 12 WMMAs, 4-way accumulator interleave: hi*hi, hi*lo, lo*hi
    BF_WMMA(acc0, ahi0, bhi) BF_WMMA(acc1, ahi1, bhi)
    BF_WMMA(acc2, ahi2, bhi) BF_WMMA(acc3, ahi3, bhi)
    BF_WMMA(acc0, ahi0, blo) BF_WMMA(acc1, ahi1, blo)
    BF_WMMA(acc2, ahi2, blo) BF_WMMA(acc3, ahi3, blo)
    BF_WMMA(acc0, alo0, bhi) BF_WMMA(acc1, alo1, bhi)
    BF_WMMA(acc2, alo2, bhi) BF_WMMA(acc3, alo3, bhi)
  }

  // ---- Epilogue: tanh + skip (skip reconstructed from hi+lo in LDS) ----
#pragma unroll
  for (int mt = 0; mt < BPG; ++mt) {
    if (b0 + mt < B) {
      v8f acc = (mt == 0) ? acc0 : (mt == 1) ? acc1 : (mt == 2) ? acc2 : acc3;
#pragma unroll
      for (int i = 0; i < 8; ++i) {
        int M = i + 8 * lh;
        int row = mt * NPART + M;
        int off = row * SB + gcol;
        float skip = bf2f(Xhi[off]) + bf2f(Xlo[off]);
        out[((size_t)b0 * NPART + row) * ND + gcol] = tanhf(acc[i]) + skip;
      }
    }
  }
}

// Weight prep: transpose [Wu;Wm;W2] (832x256 f32) -> K-contiguous bf16 hi/lo
__global__ __launch_bounds__(256) void ferminet_wprep(
    const float* __restrict__ Wu, const float* __restrict__ Wm,
    const float* __restrict__ W2, unsigned short* __restrict__ wt_hi,
    unsigned short* __restrict__ wt_lo) {
  int idx = blockIdx.x * 256 + threadIdx.x;  // 0 .. 256*832-1
  int n = idx & 255;
  int k = idx >> 8;
  if (k >= KTOT) return;
  float x;
  if (k < 256)      x = Wu[(size_t)k * ND + n];
  else if (k < 768) x = Wm[(size_t)(k - 256) * ND + n];
  else              x = W2[(size_t)(k - 768) * ND + n];
  unsigned short h, l;
  split2(x, h, l);
  wt_hi[(size_t)n * KTOT + k] = h;
  wt_lo[(size_t)n * KTOT + k] = l;
}

// =====================================================================
// f32 WMMA fallback (used only if ws_size is too small for weight prep)
// =====================================================================
#define STRIDE 836

#define WMMA_STEP(KK)                                                          \
  {                                                                            \
    v2f bf;                                                                    \
    bf.x = wp[(size_t)(k)*ND];                                                 \
    bf.y = wp[(size_t)(k + 1) * ND];                                           \
    v2f A0 = *(const v2f*)(&lds[a0 + (KK)]);                                   \
    v2f A1 = *(const v2f*)(&lds[a1 + (KK)]);                                   \
    v2f A2 = *(const v2f*)(&lds[a2 + (KK)]);                                   \
    v2f A3 = *(const v2f*)(&lds[a3 + (KK)]);                                   \
    acc0 = __builtin_amdgcn_wmma_f32_16x16x4_f32(false, A0, false, bf,         \
                                                 (short)0, acc0, false, false);\
    acc1 = __builtin_amdgcn_wmma_f32_16x16x4_f32(false, A1, false, bf,         \
                                                 (short)0, acc1, false, false);\
    acc2 = __builtin_amdgcn_wmma_f32_16x16x4_f32(false, A2, false, bf,         \
                                                 (short)0, acc2, false, false);\
    acc3 = __builtin_amdgcn_wmma_f32_16x16x4_f32(false, A3, false, bf,         \
                                                 (short)0, acc3, false, false);\
  }

#define GEMM_SEGMENT(WSEG, KS, KBASE)                                          \
  {                                                                            \
    const float* wp = (WSEG) + (size_t)(2 * lh) * ND + gcol;                   \
    _Pragma("unroll 4") for (int k = 0; k < (KS); k += 4) {                    \
      WMMA_STEP((KBASE) + k)                                                   \
    }                                                                          \
  }

__global__ __launch_bounds__(NTHREADS) void ferminet_f32(
    const float* __restrict__ in_1e, const float* __restrict__ in_2e,
    const float* __restrict__ Wu, const float* __restrict__ bu,
    const float* __restrict__ Wm, const float* __restrict__ W2,
    float* __restrict__ out, int B) {
  extern __shared__ float lds[];

  const int tid = threadIdx.x;
  const int lane = tid & 31;
  const int wid = tid >> 5;
  const int b0 = blockIdx.x * BPG;

  {
#pragma unroll
    for (int it = 0; it < (MROWS * D1) / (NTHREADS * 4); ++it) {
      int idx = (it * NTHREADS + tid) * 4;
      int row = idx >> 8;
      int col = idx & 255;
      int gb = b0 + (row >> 4);
      if (gb > B - 1) gb = B - 1;
      float4 v = *reinterpret_cast<const float4*>(
          in_1e + ((size_t)gb * NPART + (row & 15)) * D1 + col);
      *reinterpret_cast<float4*>(&lds[row * STRIDE + col]) = v;
    }
  }
  __syncthreads();

  {
#pragma unroll
    for (int w = 0; w < 4; ++w) {
      int idx = tid + w * NTHREADS;
      int lb = idx >> 9;
      int rem = idx & 511;
      int t = rem >> 8;
      int d = rem & 255;
      int base = (lb * NPART + t * 8) * STRIDE + d;
      float s = 0.f;
#pragma unroll
      for (int r = 0; r < 8; ++r) s += lds[base + r * STRIDE];
      s *= 0.125f;
      int rows_same = (lb * NPART + t * 8) * STRIDE;
      int rows_oth = (lb * NPART + (1 - t) * 8) * STRIDE;
#pragma unroll
      for (int r = 0; r < 8; ++r) {
        lds[rows_same + r * STRIDE + 256 + d] = s;
        lds[rows_oth + r * STRIDE + 512 + d] = s;
      }
    }
  }

  {
#pragma unroll
    for (int w = 0; w < 8; ++w) {
      int c = wid + w * 16;
      int lb = c >> 5;
      int m = (c >> 1) & 15;
      int j = c & 1;
      int gb = b0 + lb;
      if (gb > B - 1) gb = B - 1;
      const float* p =
          in_2e + (((size_t)gb * NPART + m) * NPART + j * 8) * D2 + lane;
      float s = 0.f;
#pragma unroll
      for (int cc = 0; cc < 8; ++cc) s += p[cc * D2];
      s *= 0.125f;
      int srow = lb * NPART + m;
      int sspin = m >> 3;
      lds[srow * STRIDE + 768 + ((j ^ sspin) << 5) + lane] = s;
    }
  }
  __syncthreads();

  const int lh = lane >> 4;
  const int ln = lane & 15;
  const int gcol = (wid << 4) + ln;

  const float bias = bu[gcol];
  v8f acc0, acc1, acc2, acc3;
#pragma unroll
  for (int i = 0; i < 8; ++i) {
    acc0[i] = bias; acc1[i] = bias; acc2[i] = bias; acc3[i] = bias;
  }

  const int a0 = (0 * 16 + ln) * STRIDE + 2 * lh;
  const int a1 = a0 + 16 * STRIDE;
  const int a2 = a1 + 16 * STRIDE;
  const int a3 = a2 + 16 * STRIDE;

  GEMM_SEGMENT(Wu, 256, 0)
  GEMM_SEGMENT(Wm, 512, 256)
  GEMM_SEGMENT(W2, 64, 768)

#pragma unroll
  for (int mt = 0; mt < BPG; ++mt) {
    if (b0 + mt < B) {
      v8f acc = (mt == 0) ? acc0 : (mt == 1) ? acc1 : (mt == 2) ? acc2 : acc3;
#pragma unroll
      for (int i = 0; i < 8; ++i) {
        int M = i + 8 * lh;
        int row = mt * NPART + M;
        float v = tanhf(acc[i]) + lds[row * STRIDE + gcol];
        out[((size_t)b0 * NPART + row) * ND + gcol] = v;
      }
    }
  }
}

// =====================================================================
extern "C" void kernel_launch(void* const* d_in, const int* in_sizes, int n_in,
                              void* d_out, int out_size, void* d_ws,
                              size_t ws_size, hipStream_t stream) {
  const float* in_1e = (const float*)d_in[0];
  const float* in_2e = (const float*)d_in[1];
  const float* Wu = (const float*)d_in[2];
  const float* bu = (const float*)d_in[3];
  const float* Wm = (const float*)d_in[4];
  const float* W2 = (const float*)d_in[5];
  float* out = (float*)d_out;

  const int B = in_sizes[0] / (NPART * D1);  // 4096
  const int grid = (B + BPG - 1) / BPG;      // 1024

  const size_t WS_NEED = (size_t)2 * KTOT * ND * sizeof(unsigned short);  // 852KB

  if (ws_size >= WS_NEED) {
    unsigned short* wt_hi = (unsigned short*)d_ws;
    unsigned short* wt_lo = wt_hi + (size_t)KTOT * ND;
    ferminet_wprep<<<(KTOT * ND + 255) / 256, 256, 0, stream>>>(Wu, Wm, W2,
                                                                wt_hi, wt_lo);
    const size_t shmem = (size_t)2 * MROWS * SB * sizeof(unsigned short);
    (void)hipFuncSetAttribute(reinterpret_cast<const void*>(ferminet_bf16),
                              hipFuncAttributeMaxDynamicSharedMemorySize,
                              (int)shmem);
    ferminet_bf16<<<grid, NTHREADS, shmem, stream>>>(in_1e, in_2e, bu, wt_hi,
                                                     wt_lo, out, B);
  } else {
    const size_t shmem = (size_t)MROWS * STRIDE * sizeof(float);
    (void)hipFuncSetAttribute(reinterpret_cast<const void*>(ferminet_f32),
                              hipFuncAttributeMaxDynamicSharedMemorySize,
                              (int)shmem);
    ferminet_f32<<<grid, NTHREADS, shmem, stream>>>(in_1e, in_2e, Wu, bu, Wm,
                                                    W2, out, B);
  }
}
